// FlowNet3DFeature_1872605741579
// MI455X (gfx1250) — matrix operand order repaired and checked by reading, not compiled, not run on hardware
//
#include <hip/hip_runtime.h>
#include <math.h>

typedef __attribute__((ext_vector_type(16))) _Float16 v16h;
typedef __attribute__((ext_vector_type(8)))  _Float16 v8h;
typedef __attribute__((ext_vector_type(8)))  float    v8f;

#define BATCH 4

static inline int ceil_div_i(int a, int b) { return (a + b - 1) / b; }

// ---------------- pc2 [B,3,N] -> xyz [B,N,3] ----------------
__global__ void k_transpose(const float* __restrict__ src, float* __restrict__ dst, int N) {
  int t = blockIdx.x * blockDim.x + threadIdx.x;
  if (t >= BATCH * N) return;
  int b = t / N, n = t % N;
  dst[(size_t)t * 3 + 0] = src[((size_t)b * 3 + 0) * N + n];
  dst[(size_t)t * 3 + 1] = src[((size_t)b * 3 + 1) * N + n];
  dst[(size_t)t * 3 + 2] = src[((size_t)b * 3 + 2) * N + n];
}

// ---------------- farthest point sampling, 1 block per batch ----------------
// dist array lives in LDS (max N=8192 -> 32KB of the 320KB WGP LDS).
__global__ void k_fps(const float* __restrict__ xyz, int N, int S, float* __restrict__ newXyz) {
  __shared__ float s_dist[8192];
  __shared__ float s_rv[256];
  __shared__ int   s_ri[256];
  __shared__ int   s_sel[1024];
  int b = blockIdx.x, tid = threadIdx.x, nt = blockDim.x;
  const float* base = xyz + (size_t)b * N * 3;
  for (int n = tid; n < N; n += nt) s_dist[n] = 1e10f;
  __syncthreads();
  int far = 0;
  for (int i = 0; i < S; i++) {
    if (tid == 0) s_sel[i] = far;
    float cx = base[far * 3 + 0], cy = base[far * 3 + 1], cz = base[far * 3 + 2];
    float best = -1.0f; int bi = 0;
    for (int n = tid; n < N; n += nt) {
      float dx = base[n * 3 + 0] - cx;
      float dy = base[n * 3 + 1] - cy;
      float dz = base[n * 3 + 2] - cz;
      float d = dx * dx + dy * dy + dz * dz;
      float dd = fminf(s_dist[n], d);
      s_dist[n] = dd;
      if (dd > best) { best = dd; bi = n; }   // strict > keeps smallest index (argmax tie rule)
    }
    s_rv[tid] = best; s_ri[tid] = bi;
    __syncthreads();
    for (int off = nt >> 1; off > 0; off >>= 1) {
      if (tid < off) {
        float ov = s_rv[tid + off]; int oi = s_ri[tid + off];
        if (ov > s_rv[tid] || (ov == s_rv[tid] && oi < s_ri[tid])) { s_rv[tid] = ov; s_ri[tid] = oi; }
      }
      __syncthreads();
    }
    far = s_ri[0];
    __syncthreads();
  }
  for (int i = tid; i < S; i += nt) {
    int id = s_sel[i];
    newXyz[((size_t)b * S + i) * 3 + 0] = base[id * 3 + 0];
    newXyz[((size_t)b * S + i) * 3 + 1] = base[id * 3 + 1];
    newXyz[((size_t)b * S + i) * 3 + 2] = base[id * 3 + 2];
  }
}

// ---------------- ball query: first K ascending hits within r, pad with first hit ----------------
__global__ void k_ballquery(const float* __restrict__ xyz, const float* __restrict__ nxyz,
                            int N, int S, int K, float r2, int* __restrict__ gidx) {
  int t = blockIdx.x * blockDim.x + threadIdx.x;
  if (t >= BATCH * S) return;
  int b = t / S;
  const float* base = xyz + (size_t)b * N * 3;
  float cx = nxyz[(size_t)t * 3 + 0], cy = nxyz[(size_t)t * 3 + 1], cz = nxyz[(size_t)t * 3 + 2];
  int cnt = 0, first = 0; bool havefirst = false;
  for (int n = 0; n < N && cnt < K; n++) {
    float dx = base[n * 3 + 0] - cx;
    float dy = base[n * 3 + 1] - cy;
    float dz = base[n * 3 + 2] - cz;
    float d = dx * dx + dy * dy + dz * dz;
    if (d <= r2) {
      if (!havefirst) { first = n; havefirst = true; }
      gidx[(size_t)t * K + cnt] = n; cnt++;
    }
  }
  for (; cnt < K; cnt++) gidx[(size_t)t * K + cnt] = first;
}

// ---------------- grouped input A[rows, LDA] (f16): [centered xyz | gathered pts | zero pad] ----------------
// pts comes either as f32 (level 1: raw xyz) or f16 (pooled features); exactly one is non-null.
__global__ void k_group(const float* __restrict__ xyz, const float* __restrict__ ptsF,
                        const _Float16* __restrict__ ptsH,
                        const float* __restrict__ nxyz, const int* __restrict__ gidx,
                        int N, int S, int K, int D, int LDA, _Float16* __restrict__ A) {
  size_t t = (size_t)blockIdx.x * blockDim.x + threadIdx.x;
  size_t total = (size_t)BATCH * S * K * LDA;
  if (t >= total) return;
  int c = (int)(t % LDA);
  size_t r = t / LDA;
  int k = (int)(r % K);
  int bs = (int)(r / K);
  int b = bs / S;
  int n = gidx[(size_t)bs * K + k];
  float v = 0.0f;
  if (c < 3)          v = xyz[((size_t)b * N + n) * 3 + c] - nxyz[(size_t)bs * 3 + c];
  else if (c < 3 + D) v = ptsF ? ptsF[((size_t)b * N + n) * D + (c - 3)]
                               : (float)ptsH[((size_t)b * N + n) * D + (c - 3)];
  A[(size_t)r * LDA + c] = (_Float16)v;
}

// ---------------- pad + f32->f16 weight staging: W[Cout,Cin] -> Wh[Cout,LDK] ----------------
__global__ void k_wpad(const float* __restrict__ Wf, _Float16* __restrict__ Wh,
                       int Cout, int Cin, int LDK) {
  int t = blockIdx.x * blockDim.x + threadIdx.x;
  if (t >= Cout * LDK) return;
  int k = t % LDK, o = t / LDK;
  Wh[t] = (k < Cin) ? (_Float16)Wf[(size_t)o * Cin + k] : (_Float16)0.0f;
}

// ---------------- WMMA GEMM: Out[M,N] = A[M,Kp(f16,lda)] * Wh[N,Kp(f16)]^T + bias ----------------
// One wave per 16x16 tile; v_wmma_f32_16x16x32_f16, 4 aligned b128 loads per k-step.
// CHECK_M=false for M%16==0 GEMMs (no divergent guard in the hot loop).
template <bool CHECK_M>
__global__ void k_gemm(const _Float16* __restrict__ A, int lda,
                       const _Float16* __restrict__ Wh, int ldk,
                       const float* __restrict__ bias,
                       float* __restrict__ Out, int M, int N, int Kp) {
  int lane = threadIdx.x & 31;
  int wave = blockIdx.x * (blockDim.x >> 5) + (threadIdx.x >> 5);
  int tilesN = N >> 4;
  int Mt = (M + 15) >> 4;
  if (wave >= Mt * tilesN) return;           // wave-uniform exit: EXEC stays all-1 for WMMA
  int tm = wave / tilesN, tn = wave % tilesN;
  int hi = lane >> 4;                         // lane half (0: lanes 0-15, 1: lanes 16-31)
  int l  = lane & 15;

  int arow = tm * 16 + l;
  bool rowok = !CHECK_M || (arow < M);
  // A fragment: elems 0-7 <- K = ka0..ka0+7; elems 8-15 <- K = ka0+16..ka0+23
  const v8h* Abase = (const v8h*)(A + (size_t)arow * lda + (hi ? 8 : 0));
  // B fragment: elems 0-7 <- K = ka0..ka0+7; elems 8-15 <- K = ka0+16..ka0+23
  const v8h* Wbase = (const v8h*)(Wh + (size_t)(tn * 16 + l) * ldk + (hi ? 8 : 0));

  v8f acc = {};
  for (int k0 = 0; k0 < Kp; k0 += 32) {
    int kv = k0 >> 3;                         // v8h index step
    v16h av = {};
    if (rowok) {
      v8h a0 = Abase[kv];
      v8h a1 = Abase[kv + 2];
#pragma unroll
      for (int j = 0; j < 8; j++) { av[j] = a0[j]; av[8 + j] = a1[j]; }
    }
    v8h b0 = Wbase[kv];
    v8h b1 = Wbase[kv + 2];
    v16h bv;
#pragma unroll
    for (int j = 0; j < 8; j++) { bv[j] = b0[j]; bv[8 + j] = b1[j]; }
    acc = __builtin_amdgcn_wmma_f32_16x16x32_f16(false, av, false, bv, (short)0, acc, false, false);
  }

  int bcol = tn * 16 + l;
  float bb = bias ? bias[bcol] : 0.0f;
  int rbase = tm * 16 + (hi ? 8 : 0);
#pragma unroll
  for (int v = 0; v < 8; v++) {
    int r = rbase + v;
    if (!CHECK_M || r < M) Out[(size_t)r * N + bcol] = acc[v] + bb;
  }
}

// ---------------- BN train-stats: one block per channel -> scale/shift ----------------
__global__ void k_bnstats(const float* __restrict__ X, int M, int C,
                          const float* __restrict__ gamma, const float* __restrict__ beta,
                          float* __restrict__ scale, float* __restrict__ shift) {
  __shared__ float s1[256];
  __shared__ float s2[256];
  int c = blockIdx.x, tid = threadIdx.x;
  float s = 0.0f, sq = 0.0f;
  for (int r = tid; r < M; r += blockDim.x) {
    float v = X[(size_t)r * C + c];
    s += v; sq += v * v;
  }
  s1[tid] = s; s2[tid] = sq;
  __syncthreads();
  for (int off = blockDim.x >> 1; off > 0; off >>= 1) {
    if (tid < off) { s1[tid] += s1[tid + off]; s2[tid] += s2[tid + off]; }
    __syncthreads();
  }
  if (tid == 0) {
    float mean = s1[0] / (float)M;
    float var  = s2[0] / (float)M - mean * mean;
    float inv  = rsqrtf(var + 1e-5f) * gamma[c];
    scale[c] = inv;
    shift[c] = beta[c] - mean * inv;
  }
}

// BN+ReLU, f32 preact -> f16 activation (feeds next WMMA GEMM)
__global__ void k_bnrelu16(const float* __restrict__ X, _Float16* __restrict__ Y,
                           const float* __restrict__ scale, const float* __restrict__ shift,
                           int total, int C) {
  int t = blockIdx.x * blockDim.x + threadIdx.x;
  if (t >= total) return;
  int c = t % C;
  float v = X[t] * scale[c] + shift[c];
  Y[t] = (_Float16)fmaxf(v, 0.0f);
}

// BN+ReLU in place, f32 (for f1/f2 output slices in d_out)
__global__ void k_bnrelu32(float* __restrict__ X,
                           const float* __restrict__ scale, const float* __restrict__ shift,
                           int total, int C) {
  int t = blockIdx.x * blockDim.x + threadIdx.x;
  if (t >= total) return;
  int c = t % C;
  float v = X[t] * scale[c] + shift[c];
  X[t] = fmaxf(v, 0.0f);
}

// ---------------- max over nsample K (f16 -> f16) ----------------
__global__ void k_maxk(const _Float16* __restrict__ X, int BS, int K, int C,
                       _Float16* __restrict__ Out) {
  int t = blockIdx.x * blockDim.x + threadIdx.x;
  if (t >= BS * C) return;
  int bs = t / C, c = t % C;
  const _Float16* p = X + ((size_t)bs * K) * C + c;
  float m = -1e30f;
  for (int k = 0; k < K; k++) m = fmaxf(m, (float)p[(size_t)k * C]);
  Out[t] = (_Float16)m;
}

// ---------------- max(+residual f32) + mean over S; max -> f16 head input, mean -> f32 d_out ----------------
__global__ void k_pools(const _Float16* __restrict__ F, int S, int C,
                        const float* __restrict__ addend,
                        _Float16* __restrict__ outMax, float* __restrict__ outMean) {
  int t = blockIdx.x * blockDim.x + threadIdx.x;
  if (t >= BATCH * C) return;
  int b = t / C, c = t % C;
  const _Float16* p = F + ((size_t)b * S) * C + c;
  float m = -1e30f, s = 0.0f;
  for (int i = 0; i < S; i++) {
    float v = (float)p[(size_t)i * C];
    m = fmaxf(m, v); s += v;
  }
  outMax[t]  = (_Float16)(m + (addend ? addend[t] : 0.0f));
  outMean[t] = s / (float)S;
}

// =======================================================================================
extern "C" void kernel_launch(void* const* d_in, const int* in_sizes, int n_in,
                              void* d_out, int out_size, void* d_ws, size_t ws_size,
                              hipStream_t stream) {
  (void)in_sizes; (void)n_in; (void)out_size; (void)ws_size;
  const int N0 = 8192;
  const float* pc2 = (const float*)d_in[1];   // pc1 (d_in[0]) is unused by the reference
  auto W = [&](int i) { return (const float*)d_in[i]; };

  // ---- workspace layout (256B aligned slices)
  char* ws = (char*)d_ws;
  size_t off = 0;
  auto alloc = [&](size_t bytes) { void* p = ws + off; off += (bytes + 255) & ~(size_t)255; return p; };
  float*    xyz0  = (float*)alloc((size_t)BATCH * N0 * 3 * 4);
  float*    nx1   = (float*)alloc((size_t)BATCH * 1024 * 3 * 4);
  float*    nx2   = (float*)alloc((size_t)BATCH * 256 * 3 * 4);
  float*    nx3   = (float*)alloc((size_t)BATCH * 64 * 3 * 4);
  int*      gidx  = (int*)  alloc((size_t)BATCH * 1024 * 16 * 4);
  float*    preX  = (float*)alloc((size_t)4194304 * 4);        // f32 pre-activations (max 65536x64)
  _Float16* actA  = (_Float16*)alloc((size_t)4194304 * 2);     // f16 activations ping
  _Float16* actB  = (_Float16*)alloc((size_t)4194304 * 2);     // f16 activations pong
  _Float16* feat1 = (_Float16*)alloc((size_t)BATCH * 1024 * 64 * 2);
  _Float16* feat2 = (_Float16*)alloc((size_t)BATCH * 256 * 256 * 2);
  _Float16* feat3 = (_Float16*)alloc((size_t)BATCH * 64 * 1024 * 2);
  _Float16* hIn   = (_Float16*)alloc((size_t)BATCH * 1024 * 2);  // head input (pool max + residual)
  _Float16* hMid  = (_Float16*)alloc((size_t)BATCH * 2048 * 2);  // head mid activation
  float*    scl   = (float*)alloc(4096 * 4);
  float*    shf   = (float*)alloc(4096 * 4);
  _Float16* wh    = (_Float16*)alloc((size_t)4096 * 2048 * 2);

  // ---- output slices (tuple flattened: out, f2, f1, mean_l3, mean_l2, mean_l1)
  float* out = (float*)d_out;
  float* f2o = out + 16384;
  float* f1o = out + 20480;
  float* ml3 = out + 21504;
  float* ml2 = out + 25600;
  float* ml1 = out + 26624;

  auto gemm = [&](const _Float16* A, int lda, const float* Wf, const float* bias,
                  float* O, int M, int Nc, int Cin) {
    int ldk = lda;  // activations are padded so Kp == lda
    int wtot = Nc * ldk;
    k_wpad<<<ceil_div_i(wtot, 256), 256, 0, stream>>>(Wf, wh, Nc, Cin, ldk);
    int tiles = ((M + 15) / 16) * (Nc / 16);
    if ((M & 15) == 0)
      k_gemm<false><<<ceil_div_i(tiles, 8), 256, 0, stream>>>(A, lda, wh, ldk, bias, O, M, Nc, lda);
    else
      k_gemm<true><<<ceil_div_i(tiles, 8), 256, 0, stream>>>(A, lda, wh, ldk, bias, O, M, Nc, lda);
  };
  auto bn16 = [&](const float* X, _Float16* Y, int M, int C,
                  const float* gamma, const float* beta) {
    k_bnstats<<<C, 256, 0, stream>>>(X, M, C, gamma, beta, scl, shf);
    int tot = M * C;
    k_bnrelu16<<<ceil_div_i(tot, 256), 256, 0, stream>>>(X, Y, scl, shf, tot, C);
  };
  auto bn32 = [&](float* X, int M, int C, const float* gamma, const float* beta) {
    k_bnstats<<<C, 256, 0, stream>>>(X, M, C, gamma, beta, scl, shf);
    int tot = M * C;
    k_bnrelu32<<<ceil_div_i(tot, 256), 256, 0, stream>>>(X, scl, shf, tot, C);
  };

  // ---- stage 0
  k_transpose<<<ceil_div_i(BATCH * N0, 256), 256, 0, stream>>>(pc2, xyz0, N0);

  // ---- SA1: N=8192, S=1024, K=16, r=0.5, C0=6 (pad 32)
  k_fps<<<BATCH, 256, 0, stream>>>(xyz0, N0, 1024, nx1);
  k_ballquery<<<ceil_div_i(BATCH * 1024, 128), 128, 0, stream>>>(xyz0, nx1, N0, 1024, 16, 0.25f, gidx);
  k_group<<<ceil_div_i(BATCH * 1024 * 16 * 32, 256), 256, 0, stream>>>(
      xyz0, xyz0, nullptr, nx1, gidx, N0, 1024, 16, 3, 32, actA);
  int M1 = BATCH * 1024 * 16;
  gemm(actA, 32, W(5),  W(2),  preX, M1, 32, 6);   bn16(preX, actB, M1, 32,  W(4),  W(3));
  gemm(actB, 32, W(9),  W(6),  preX, M1, 64, 32);  bn16(preX, actA, M1, 64,  W(8),  W(7));
  gemm(actA, 64, W(13), W(10), preX, M1, 64, 64);  bn16(preX, actB, M1, 64,  W(12), W(11));
  k_maxk<<<ceil_div_i(BATCH * 1024 * 64, 256), 256, 0, stream>>>(actB, BATCH * 1024, 16, 64, feat1);
  k_pools<<<ceil_div_i(BATCH * 64, 64), 64, 0, stream>>>(feat1, 1024, 64, nullptr, hIn, ml1);

  // ---- head1: [4,64] -> 128 -> f1 [4,256]
  gemm(hIn, 64,   W(41), W(38), preX, 4, 128, 64);  bn16(preX, hMid, 4, 128, W(40), W(39));
  gemm(hMid, 128, W(45), W(42), f1o,  4, 256, 128); bn32(f1o, 4, 256, W(44), W(43));

  // ---- SA2: N=1024, S=256, K=16, r=1.0, C0=67 (pad 96)
  k_fps<<<BATCH, 256, 0, stream>>>(nx1, 1024, 256, nx2);
  k_ballquery<<<ceil_div_i(BATCH * 256, 128), 128, 0, stream>>>(nx1, nx2, 1024, 256, 16, 1.0f, gidx);
  k_group<<<ceil_div_i(BATCH * 256 * 16 * 96, 256), 256, 0, stream>>>(
      nx1, nullptr, feat1, nx2, gidx, 1024, 256, 16, 64, 96, actA);
  int M2 = BATCH * 256 * 16;
  gemm(actA, 96,  W(17), W(14), preX, M2, 64, 67);   bn16(preX, actB, M2, 64,  W(16), W(15));
  gemm(actB, 64,  W(21), W(18), preX, M2, 128, 64);  bn16(preX, actA, M2, 128, W(20), W(19));
  gemm(actA, 128, W(25), W(22), preX, M2, 256, 128); bn16(preX, actB, M2, 256, W(24), W(23));
  k_maxk<<<ceil_div_i(BATCH * 256 * 256, 256), 256, 0, stream>>>(actB, BATCH * 256, 16, 256, feat2);
  k_pools<<<ceil_div_i(BATCH * 256, 64), 64, 0, stream>>>(feat2, 256, 256, f1o, hIn, ml2);

  // ---- head2: [4,256] -> 512 -> f2 [4,1024]
  gemm(hIn, 256,  W(49), W(46), preX, 4, 512, 256);  bn16(preX, hMid, 4, 512, W(48), W(47));
  gemm(hMid, 512, W(53), W(50), f2o,  4, 1024, 512); bn32(f2o, 4, 1024, W(52), W(51));

  // ---- SA3: N=256, S=64, K=8, r=2.0, C0=259 (pad 288)
  k_fps<<<BATCH, 256, 0, stream>>>(nx2, 256, 64, nx3);
  k_ballquery<<<ceil_div_i(BATCH * 64, 64), 64, 0, stream>>>(nx2, nx3, 256, 64, 8, 4.0f, gidx);
  k_group<<<ceil_div_i(BATCH * 64 * 8 * 288, 256), 256, 0, stream>>>(
      nx2, nullptr, feat2, nx3, gidx, 256, 64, 8, 256, 288, actA);
  int M3 = BATCH * 64 * 8;
  gemm(actA, 288, W(29), W(26), preX, M3, 256, 259);  bn16(preX, actB, M3, 256,  W(28), W(27));
  gemm(actB, 256, W(33), W(30), preX, M3, 512, 256);  bn16(preX, actA, M3, 512,  W(32), W(31));
  gemm(actA, 512, W(37), W(34), preX, M3, 1024, 512); bn16(preX, actB, M3, 1024, W(36), W(35));
  k_maxk<<<ceil_div_i(BATCH * 64 * 1024, 256), 256, 0, stream>>>(actB, BATCH * 64, 8, 1024, feat3);
  k_pools<<<ceil_div_i(BATCH * 1024, 64), 64, 0, stream>>>(feat3, 64, 1024, f2o, hIn, ml3);

  // ---- head3: [4,1024] -> 2048 -> out [4,4096] (final layer: plain Linear)
  gemm(hIn, 1024,  W(57), W(54), preX, 4, 2048, 1024); bn16(preX, hMid, 4, 2048, W(56), W(55));
  gemm(hMid, 2048, W(59), W(58), out,  4, 4096, 2048);
}
